// BingramLanguageModel_89824946028662
// MI455X (gfx1250) — matrix operand order repaired and checked by reading, not compiled, not run on hardware
//
#include <hip/hip_runtime.h>
#include <hip/hip_bf16.h>
#include <stdint.h>

typedef __bf16 bf16;
typedef bf16  v16bf __attribute__((ext_vector_type(16)));
typedef bf16  v8bf  __attribute__((ext_vector_type(8)));
typedef float v8f   __attribute__((ext_vector_type(8)));

// Model dimensions (fixed by the reference)
constexpr int B_  = 2;
constexpr int T_  = 1024;
constexpr int E_  = 768;
constexpr int H_  = 12;
constexpr int L_  = 6;
constexpr int V_  = 32000;
constexpr int Dh_ = E_ / H_;   // 64
constexpr int DF_ = 4 * E_;    // 3072
constexpr int M_  = B_ * T_;   // 2048 rows through every GEMM

// GEMM blocking
constexpr int BM = 128;        // block tile M
constexpr int BN = 128;        // block tile N
constexpr int BK = 32;         // K step (one bf16 WMMA)
constexpr int SPAD = 48;       // LDS row stride in halfs (96B: 32B-aligned, bank-spread)

// ---------------------------------------------------------------------------
// CDNA5 async global->LDS copy (ASYNCcnt-tracked, bypasses VGPRs).
// VDST VGPR = LDS byte address (low 32 bits of the generic shared pointer),
// VADDR = 64-bit global address, saddr = off.
// ---------------------------------------------------------------------------
__device__ __forceinline__ void async_copy_b128(uint32_t lds_addr,
                                                const void* gptr) {
  asm volatile("global_load_async_to_lds_b128 %0, %1, off"
               :
               : "v"(lds_addr), "v"(gptr)
               : "memory");
}
__device__ __forceinline__ void wait_async0() {
  asm volatile("s_wait_asynccnt 0" ::: "memory");
}

// ---------------------------------------------------------------------------
// f32 -> bf16 conversion (weights); grid-stride
// ---------------------------------------------------------------------------
__global__ void f32_to_bf16_kernel(const float* __restrict__ in,
                                   bf16* __restrict__ out, size_t n) {
  size_t i = (size_t)blockIdx.x * blockDim.x + threadIdx.x;
  size_t stride = (size_t)gridDim.x * blockDim.x;
  for (; i < n; i += stride) out[i] = (bf16)in[i];
}

// ---------------------------------------------------------------------------
// Embedding: x[b,t,:] = tok_emb[idx[b,t],:] + pos_emb[t,:]
// ---------------------------------------------------------------------------
__global__ __launch_bounds__(256) void embed_kernel(
    const int* __restrict__ idx, const float* __restrict__ tok,
    const float* __restrict__ pos, float* __restrict__ x) {
  const int row = blockIdx.x;  // b*T + t
  const int t = row % T_;
  const int token = idx[row];
  const float* tr = tok + (size_t)token * E_;
  const float* pr = pos + (size_t)t * E_;
  float* xr = x + (size_t)row * E_;
  for (int e = threadIdx.x; e < E_; e += 256) xr[e] = tr[e] + pr[e];
}

// ---------------------------------------------------------------------------
// LayerNorm over E per row, emit bf16 activations for the next GEMM
// ---------------------------------------------------------------------------
__global__ __launch_bounds__(256) void layernorm_bf16_kernel(
    const float* __restrict__ x, const float* __restrict__ g,
    const float* __restrict__ bsrc, bf16* __restrict__ out, int n) {
  __shared__ float red[8];
  __shared__ float stats[2];
  const int row = blockIdx.x;
  const float* xr = x + (size_t)row * n;

  float s = 0.f;
  for (int e = threadIdx.x; e < n; e += 256) s += xr[e];
#pragma unroll
  for (int o = 16; o > 0; o >>= 1) s += __shfl_down(s, o, 32);
  if ((threadIdx.x & 31) == 0) red[threadIdx.x >> 5] = s;
  __syncthreads();
  if (threadIdx.x == 0) {
    float t = 0.f;
#pragma unroll
    for (int i = 0; i < 8; ++i) t += red[i];
    stats[0] = t / n;
  }
  __syncthreads();
  const float mean = stats[0];

  float vs = 0.f;
  for (int e = threadIdx.x; e < n; e += 256) {
    float d = xr[e] - mean;
    vs += d * d;
  }
#pragma unroll
  for (int o = 16; o > 0; o >>= 1) vs += __shfl_down(vs, o, 32);
  if ((threadIdx.x & 31) == 0) red[threadIdx.x >> 5] = vs;
  __syncthreads();
  if (threadIdx.x == 0) {
    float t = 0.f;
#pragma unroll
    for (int i = 0; i < 8; ++i) t += red[i];
    stats[1] = rsqrtf(t / n + 1e-5f);
  }
  __syncthreads();
  const float rstd = stats[1];

  bf16* orow = out + (size_t)row * n;
  for (int e = threadIdx.x; e < n; e += 256)
    orow[e] = (bf16)((xr[e] - mean) * rstd * g[e] + bsrc[e]);
}

// ---------------------------------------------------------------------------
// LDS-staged WMMA GEMM: out[M,N] = A[M,K] @ W[N,K]^T (+bias,+residual,relu)
//   block = 256 threads (8 waves), tile 128(M) x 128(N), BK=32 per stage
//   staging: GLOBAL_LOAD_ASYNC_TO_LDS_B128 into double-buffered LDS (48KB)
//   wave = 32(M) x 64(N) via 8 f32 accumulators; A-frag reused 4x, B-frag 2x
//   L2-side arithmetic intensity: 64 flop/byte
//   Requires: M % 128 == 0, N % 128 == 0, K % 32 == 0 (all call sites do)
// ---------------------------------------------------------------------------
template <bool HAS_BIAS, bool HAS_RES, bool RELU, bool OUTF>
__global__ __launch_bounds__(256) void gemm_bf16_wmma(
    const bf16* __restrict__ A, const bf16* __restrict__ W,
    const float* __restrict__ bias, const float* __restrict__ resid,
    float* __restrict__ outF, bf16* __restrict__ outB, int N, int K) {
  __shared__ __align__(32) bf16 As[2][BM * SPAD];
  __shared__ __align__(32) bf16 Bs[2][BN * SPAD];

  const int tid = threadIdx.x;
  const int lane = tid & 31;
  const int wave = tid >> 5;
  const int l15 = lane & 15;
  const int hi = lane >> 4;
  const int wm = wave >> 1;  // 0..3 : 32-row strip within block tile
  const int wn = wave & 1;   // 0..1 : 64-col strip within block tile
  const int bm = blockIdx.y * BM;
  const int bn = blockIdx.x * BN;

  // global->LDS loader mapping: each thread moves 16 halfs of A and of B
  const int ldRow = tid >> 1;        // 0..127
  const int ldCol = (tid & 1) * 16;  // 0 or 16
  const bf16* gA = A + (size_t)(bm + ldRow) * K + ldCol;
  const bf16* gB = W + (size_t)(bn + ldRow) * K + ldCol;

  // LDS byte addresses for this thread's staging slot (per buffer)
  const uint32_t aBase =
      (uint32_t)(size_t)(void*)&As[0][0] + (uint32_t)(ldRow * SPAD + ldCol) * 2;
  const uint32_t bBase =
      (uint32_t)(size_t)(void*)&Bs[0][0] + (uint32_t)(ldRow * SPAD + ldCol) * 2;
  constexpr uint32_t BUFBYTES = (uint32_t)(BM * SPAD * 2);

  v8f acc[2][4] = {};
  const int nk = K / BK;

  // stage 0
  async_copy_b128(aBase, gA);
  async_copy_b128(aBase + 16, gA + 8);
  async_copy_b128(bBase, gB);
  async_copy_b128(bBase + 16, gB + 8);
  wait_async0();
  __syncthreads();

  for (int ks = 0; ks < nk; ++ks) {
    const int cur = ks & 1;
    if (ks + 1 < nk) {
      const uint32_t nb = (uint32_t)((ks + 1) & 1) * BUFBYTES;
      const bf16* ga = gA + (ks + 1) * BK;
      const bf16* gb = gB + (ks + 1) * BK;
      async_copy_b128(aBase + nb, ga);
      async_copy_b128(aBase + nb + 16, ga + 8);
      async_copy_b128(bBase + nb, gb);
      async_copy_b128(bBase + nb + 16, gb + 8);
      if (ks + 2 < nk) {
        __builtin_prefetch(gA + (ks + 2) * BK, 0, 1);  // global_prefetch_b8
        __builtin_prefetch(gB + (ks + 2) * BK, 0, 1);
      }
    }

    // A fragments (16x32): lane row = strip + l15; 16 halfs at K = hi*16
    v16bf af[2];
#pragma unroll
    for (int mi = 0; mi < 2; ++mi) {
      const bf16* p = &As[cur][(wm * 32 + mi * 16 + l15) * SPAD + hi * 16];
      v8bf lo = *(const v8bf*)p;
      v8bf hi8 = *(const v8bf*)(p + 8);
      af[mi] = __builtin_shufflevector(lo, hi8, 0, 1, 2, 3, 4, 5, 6, 7, 8, 9,
                                       10, 11, 12, 13, 14, 15);
    }
    // B fragments (32x16): lane col = strip + l15; 16 halfs at K = hi*16
#pragma unroll
    for (int nj = 0; nj < 4; ++nj) {
      v16bf bf_ =
          *(const v16bf*)&Bs[cur][(wn * 64 + nj * 16 + l15) * SPAD + hi * 16];
#pragma unroll
      for (int mi = 0; mi < 2; ++mi)
        acc[mi][nj] = __builtin_amdgcn_wmma_f32_16x16x32_bf16(
            false, af[mi], false, bf_, (short)0, acc[mi][nj], false, false);
    }

    if (ks + 1 < nk) {
      wait_async0();     // our async writes into the other buffer are done
      __syncthreads();   // everyone's writes visible; readers of it are past
    }
  }

  // straight-line epilogue (all flags compile-time)
#pragma unroll
  for (int mi = 0; mi < 2; ++mi) {
#pragma unroll
    for (int nj = 0; nj < 4; ++nj) {
      const int col = bn + wn * 64 + nj * 16 + l15;
      float bcol = 0.f;
      if (HAS_BIAS) bcol = bias[col];
#pragma unroll
      for (int r = 0; r < 8; ++r) {
        const int row = bm + wm * 32 + mi * 16 + r + 8 * hi;
        float vv = acc[mi][nj][r] + bcol;
        if (HAS_RES) vv += resid[(size_t)row * N + col];
        if (RELU) vv = vv > 0.f ? vv : 0.f;
        if (OUTF)
          outF[(size_t)row * N + col] = vv;
        else
          outB[(size_t)row * N + col] = (bf16)vv;
      }
    }
  }
}

// ---------------------------------------------------------------------------
// Flash-style causal attention, one wave per 16-query tile per (b,h).
// q,k,v,o are bf16 in [B,T,H,Dh] layout. scores: 4 WMMAs per 32-key tile
// (K over Dh=64); probs bounce through LDS (C-layout -> A-layout);
// p@v: 4 WMMAs into a 16x64 f32 accumulator with online softmax rescaling.
// ---------------------------------------------------------------------------
__global__ __launch_bounds__(32) void attn_wmma_kernel(
    const bf16* __restrict__ q, const bf16* __restrict__ k,
    const bf16* __restrict__ v, bf16* __restrict__ o, float scale) {
  __shared__ __align__(32) bf16 pl[16 * 32];  // prob tile, row-major 16x32

  const int lane = threadIdx.x & 31;
  const int l15 = lane & 15;
  const int hi = lane >> 4;
  const int qt = blockIdx.x;
  const int bh = blockIdx.y;
  const int b = bh / H_;
  const int h = bh % H_;
  const int qi0 = qt * 16;
  const size_t base = (size_t)b * T_ * E_ + (size_t)h * Dh_;

  // q A-fragments (Dh=64 -> K windows [0,32) and [32,64))
  v16bf qa[2];
  {
    const bf16* qr = q + base + (size_t)(qi0 + l15) * E_ + hi * 16;
#pragma unroll
    for (int f = 0; f < 2; ++f) {
      v8bf lo = *(const v8bf*)(qr + f * 32);
      v8bf hi8 = *(const v8bf*)(qr + f * 32 + 8);
      qa[f] = __builtin_shufflevector(lo, hi8, 0, 1, 2, 3, 4, 5, 6, 7, 8, 9,
                                      10, 11, 12, 13, 14, 15);
    }
  }

  float mrow[8], srow[8];
#pragma unroll
  for (int r = 0; r < 8; ++r) {
    mrow[r] = -1e30f;
    srow[r] = 0.f;
  }
  v8f accO[4] = {};

  const int nkt = (qi0 + 16 + 31) >> 5;  // 32-key tiles covering causal range
  for (int kt = 0; kt < nkt; ++kt) {
    const int key0 = kt * 32;

    v8f c[2] = {};
#pragma unroll
    for (int f = 0; f < 2; ++f) {
#pragma unroll
      for (int s = 0; s < 2; ++s) {
        const bf16* kp =
            k + base + (size_t)(key0 + s * 16 + l15) * E_ + f * 32 + hi * 16;
        v16bf kb = *(const v16bf*)kp;
        c[s] = __builtin_amdgcn_wmma_f32_16x16x32_bf16(
            false, qa[f], false, kb, (short)0, c[s], false, false);
      }
    }

    // scale, causal mask, online softmax; emit bf16 probs to LDS
#pragma unroll
    for (int r = 0; r < 8; ++r) {
      const int row = qi0 + r + 8 * hi;
      float v0 = c[0][r] * scale;
      float v1 = c[1][r] * scale;
      if (key0 + l15 > row) v0 = -1e30f;
      if (key0 + 16 + l15 > row) v1 = -1e30f;
      float mx = fmaxf(v0, v1);
#pragma unroll
      for (int off = 1; off < 16; off <<= 1)
        mx = fmaxf(mx, __shfl_xor(mx, off, 32));
      const float mnew = fmaxf(mrow[r], mx);
      const float alpha = __expf(mrow[r] - mnew);
      const float p0 = __expf(v0 - mnew);
      const float p1 = __expf(v1 - mnew);
      float ps = p0 + p1;
#pragma unroll
      for (int off = 1; off < 16; off <<= 1) ps += __shfl_xor(ps, off, 32);
      srow[r] = srow[r] * alpha + ps;
      mrow[r] = mnew;
#pragma unroll
      for (int j = 0; j < 4; ++j) accO[j][r] *= alpha;
      pl[(r + 8 * hi) * 32 + l15] = (bf16)p0;
      pl[(r + 8 * hi) * 32 + 16 + l15] = (bf16)p1;
    }
    __syncthreads();

    // probs: C-layout -> A-layout (16 queries x 32 keys)
    v16bf pa;
    {
      const bf16* pp = &pl[l15 * 32 + hi * 16];
      v8bf lo = *(const v8bf*)pp;
      v8bf hi8 = *(const v8bf*)(pp + 8);
      pa = __builtin_shufflevector(lo, hi8, 0, 1, 2, 3, 4, 5, 6, 7, 8, 9, 10,
                                   11, 12, 13, 14, 15);
    }
    __syncthreads();

    // accO += probs @ v  (B frag: lane col = 16j+l15, K = key 16*hi+i)
#pragma unroll
    for (int j = 0; j < 4; ++j) {
      const bf16* vp = v + base + (size_t)(key0 + hi * 16) * E_ + j * 16 + l15;
      v16bf vb;
#pragma unroll
      for (int i = 0; i < 16; ++i) vb[i] = vp[(size_t)i * E_];
      accO[j] = __builtin_amdgcn_wmma_f32_16x16x32_bf16(
          false, pa, false, vb, (short)0, accO[j], false, false);
    }
  }

  // normalize and write o (bf16, [B,T,H,Dh] layout)
#pragma unroll
  for (int j = 0; j < 4; ++j) {
#pragma unroll
    for (int r = 0; r < 8; ++r) {
      const int row = qi0 + r + 8 * hi;
      const float val = accO[j][r] / srow[r];
      o[base + (size_t)row * E_ + j * 16 + l15] = (bf16)val;
    }
  }
}

// ---------------------------------------------------------------------------
// Host-side orchestration
// ---------------------------------------------------------------------------
extern "C" void kernel_launch(void* const* d_in, const int* in_sizes, int n_in,
                              void* d_out, int out_size, void* d_ws,
                              size_t ws_size, hipStream_t stream) {
  (void)in_sizes; (void)n_in; (void)out_size; (void)ws_size;

  const int*   idx  = (const int*)d_in[0];
  const float* tok  = (const float*)d_in[1];
  const float* pos  = (const float*)d_in[2];
  const float* Wq   = (const float*)d_in[3];
  const float* Wk   = (const float*)d_in[4];
  const float* Wv   = (const float*)d_in[5];
  const float* Wo   = (const float*)d_in[6];
  const float* bo   = (const float*)d_in[7];
  const float* ln1g = (const float*)d_in[8];
  const float* ln1b = (const float*)d_in[9];
  const float* W1   = (const float*)d_in[10];
  const float* b1   = (const float*)d_in[11];
  const float* W2   = (const float*)d_in[12];
  const float* b2   = (const float*)d_in[13];
  const float* ln2g = (const float*)d_in[14];
  const float* ln2b = (const float*)d_in[15];
  const float* lnfg = (const float*)d_in[16];
  const float* lnfb = (const float*)d_in[17];
  const float* Wlm  = (const float*)d_in[18];
  const float* blm  = (const float*)d_in[19];
  float* logits = (float*)d_out;

  // Workspace carving (256B aligned)
  char* ws = (char*)d_ws;
  size_t off = 0;
  auto carve = [&](size_t bytes) -> void* {
    void* p = ws + off;
    off = (off + bytes + 255) & ~(size_t)255;
    return p;
  };
  const size_t nEE = (size_t)L_ * E_ * E_;
  const size_t nFE = (size_t)L_ * DF_ * E_;
  bf16* wq_bf  = (bf16*)carve(nEE * 2);
  bf16* wk_bf  = (bf16*)carve(nEE * 2);
  bf16* wv_bf  = (bf16*)carve(nEE * 2);
  bf16* wo_bf  = (bf16*)carve(nEE * 2);
  bf16* w1_bf  = (bf16*)carve(nFE * 2);
  bf16* w2_bf  = (bf16*)carve(nFE * 2);
  bf16* wlm_bf = (bf16*)carve((size_t)V_ * E_ * 2);
  float* x     = (float*)carve((size_t)M_ * E_ * 4);
  bf16* h_bf   = (bf16*)carve((size_t)M_ * E_ * 2);
  bf16* q_bf   = (bf16*)carve((size_t)M_ * E_ * 2);
  bf16* k_bf   = (bf16*)carve((size_t)M_ * E_ * 2);
  bf16* v_bf   = (bf16*)carve((size_t)M_ * E_ * 2);
  bf16* o_bf   = (bf16*)carve((size_t)M_ * E_ * 2);
  bf16* f1_bf  = (bf16*)carve((size_t)M_ * DF_ * 2);

  auto conv = [&](const float* src, bf16* dst, size_t n) {
    f32_to_bf16_kernel<<<2048, 256, 0, stream>>>(src, dst, n);
  };
  conv(Wq, wq_bf, nEE);
  conv(Wk, wk_bf, nEE);
  conv(Wv, wv_bf, nEE);
  conv(Wo, wo_bf, nEE);
  conv(W1, w1_bf, nFE);
  conv(W2, w2_bf, nFE);
  conv(Wlm, wlm_bf, (size_t)V_ * E_);

  // Embedding
  embed_kernel<<<M_, 256, 0, stream>>>(idx, tok, pos, x);

  const float scale = 1.0f / sqrtf((float)E_);  // reference uses E^-0.5

  for (int l = 0; l < L_; ++l) {
    const size_t wofs = (size_t)l * E_ * E_;
    const size_t fofs = (size_t)l * DF_ * E_;

    // LN1 -> h
    layernorm_bf16_kernel<<<M_, 256, 0, stream>>>(
        x, ln1g + (size_t)l * E_, ln1b + (size_t)l * E_, h_bf, E_);

    // q,k,v projections: plain GEMM, bf16 out
    {
      dim3 g(E_ / BN, M_ / BM);
      gemm_bf16_wmma<false, false, false, false><<<g, 256, 0, stream>>>(
          h_bf, wq_bf + wofs, nullptr, nullptr, nullptr, q_bf, E_, E_);
      gemm_bf16_wmma<false, false, false, false><<<g, 256, 0, stream>>>(
          h_bf, wk_bf + wofs, nullptr, nullptr, nullptr, k_bf, E_, E_);
      gemm_bf16_wmma<false, false, false, false><<<g, 256, 0, stream>>>(
          h_bf, wv_bf + wofs, nullptr, nullptr, nullptr, v_bf, E_, E_);
    }

    // causal attention
    attn_wmma_kernel<<<dim3(T_ / 16, B_ * H_), 32, 0, stream>>>(
        q_bf, k_bf, v_bf, o_bf, scale);

    // x = x + o @ Wo^T + bo : bias + residual, f32 out
    {
      dim3 g(E_ / BN, M_ / BM);
      gemm_bf16_wmma<true, true, false, true><<<g, 256, 0, stream>>>(
          o_bf, wo_bf + wofs, bo + (size_t)l * E_, x, x, nullptr, E_, E_);
    }

    // LN2 -> h2
    layernorm_bf16_kernel<<<M_, 256, 0, stream>>>(
        x, ln2g + (size_t)l * E_, ln2b + (size_t)l * E_, h_bf, E_);

    // f1 = relu(h2 @ W1^T + b1) : bias + relu, bf16 out
    {
      dim3 g(DF_ / BN, M_ / BM);
      gemm_bf16_wmma<true, false, true, false><<<g, 256, 0, stream>>>(
          h_bf, w1_bf + fofs, b1 + (size_t)l * DF_, nullptr, nullptr, f1_bf,
          DF_, E_);
    }
    // x = x + f1 @ W2^T + b2 : bias + residual, f32 out
    {
      dim3 g(E_ / BN, M_ / BM);
      gemm_bf16_wmma<true, true, false, true><<<g, 256, 0, stream>>>(
          f1_bf, w2_bf + fofs, b2 + (size_t)l * E_, x, x, nullptr, E_, DF_);
    }
  }

  // final LN + logits (bias, f32 out)
  layernorm_bf16_kernel<<<M_, 256, 0, stream>>>(x, lnfg, lnfb, h_bf, E_);
  {
    dim3 g(V_ / BN, M_ / BM);
    gemm_bf16_wmma<true, false, false, true><<<g, 256, 0, stream>>>(
        h_bf, wlm_bf, blm, nullptr, logits, nullptr, V_, E_);
  }
}
